// LiquidEchoModel_75892072121001
// MI455X (gfx1250) — matrix operand order repaired and checked by reading, not compiled, not run on hardware
//
#include <hip/hip_runtime.h>
#include <math.h>

#define VOCAB   50257
#define DMODEL  256
#define NLAYER  2
#define NOSC    512
#define NHEAD   4
#define BSZ     8
#define TSEQ    256
#define LUTN    4096

#define PHI_F      1.61803398874989484820f
#define LUT_SCALE  651.89864690440330936f   /* 4096 / (2*pi) */

typedef __attribute__((ext_vector_type(16))) _Float16 v16h;
typedef __attribute__((ext_vector_type(8)))  _Float16 v8h;
typedef __attribute__((ext_vector_type(8)))  float    v8f;

// ---------------------------------------------------------------------------
// LUT sin/cos with linear interpolation, matching reference semantics
// (floor, frac, mod-4096 indexing; &4095 == jnp.mod for int32).
// ---------------------------------------------------------------------------
__device__ __forceinline__ void lut_sincos(float theta,
                                           const float* __restrict__ st,
                                           const float* __restrict__ ct,
                                           float& s, float& c) {
    float pos  = theta * LUT_SCALE;
    float i0f  = floorf(pos);
    float frac = pos - i0f;
    int   i0   = ((int)i0f) & (LUTN - 1);
    int   i1   = (i0 + 1)   & (LUTN - 1);
    float om   = 1.0f - frac;
    s = st[i0] * om + st[i1] * frac;
    c = ct[i0] * om + ct[i1] * frac;
}

// ---------------------------------------------------------------------------
// Init: build LUT tables, zero carry state and memory.
// ---------------------------------------------------------------------------
__global__ void k_init(float* sintab, float* costab,
                       float* xr_buf, float* xi_buf,
                       float* mem_r, float* mem_i) {
    int i      = blockIdx.x * blockDim.x + threadIdx.x;
    int stride = gridDim.x * blockDim.x;
    const float step = 6.28318530717958647692f / (float)LUTN;
    for (int j = i; j < LUTN; j += stride) {
        float a = step * (float)j;
        sintab[j] = sinf(a);
        costab[j] = cosf(a);
    }
    for (int j = i; j < BSZ * DMODEL; j += stride) { xr_buf[j] = 0.f; xi_buf[j] = 0.f; }
    for (int j = i; j < NLAYER * BSZ * NHEAD * DMODEL; j += stride) { mem_r[j] = 0.f; mem_i[j] = 0.f; }
}

// ---------------------------------------------------------------------------
// theta + LUT + attn reductions: one wave32 per (b,n) row, reduce over D.
// grid = 512 blocks x 256 threads (8 waves/block) -> 4096 rows.
// ---------------------------------------------------------------------------
__global__ void k_theta_attn(const float* __restrict__ Wm,
                             const float* __restrict__ Bm,
                             const float* __restrict__ attn_cos,
                             const float* __restrict__ attn_sin,
                             const float* __restrict__ sintab,
                             const float* __restrict__ costab,
                             const float* __restrict__ xc,
                             float* __restrict__ cos_sum,
                             float* __restrict__ sin_sum,
                             int l, int t) {
    int lane = threadIdx.x & 31;
    int wave = threadIdx.x >> 5;
    int row  = blockIdx.x * 8 + wave;       // 0..4095
    int b    = row >> 9;
    int n    = row & (NOSC - 1);
    float tf = (float)t;
    size_t base = ((size_t)l * NOSC + n) * DMODEL;
    const float* wrow = Wm       + base;
    const float* brow = Bm       + base;
    const float* crow = attn_cos + base;
    const float* srow = attn_sin + base;
    const float* xrow = xc + b * DMODEL;
    float cs = 0.f, ss = 0.f;
    for (int d = lane; d < DMODEL; d += 32) {
        float th = xrow[d] / (1.f + fabsf(wrow[d])) + brow[d] + tf;
        float s, c;
        lut_sincos(th, sintab, costab, s, c);
        cs += c * crow[d];
        ss += s * srow[d];
    }
    #pragma unroll
    for (int m = 16; m >= 1; m >>= 1) {
        cs += __shfl_xor(cs, m, 32);
        ss += __shfl_xor(ss, m, 32);
    }
    if (lane == 0) {
        cos_sum[b * NOSC + n] = cs;
        sin_sum[b * NOSC + n] = ss;
    }
}

// ---------------------------------------------------------------------------
// Fused per-batch-block kernel (grid = BSZ blocks, 256 threads).
// Optionally:
//  1) update for layer l_u at time t_u (res GEMV + silu, blend/mem/echo)
//  2) write f16 Apack row for the logits GEMM
//  3) evolve state -> x for time t_n (embedding + LUT)
//  4) collapse xc for layer l_n at time t_n
//  5) trigger -> alpha for layer l_n at time t_n
// All phases for batch b are block-local, so they chain inside one kernel.
// ---------------------------------------------------------------------------
__global__ __launch_bounds__(256) void k_fused(
        const float* __restrict__ out_r, const float* __restrict__ out_i,
        const float* __restrict__ w_state, const float* __restrict__ b_state,
        const float* __restrict__ res_scale, const float* __restrict__ echo_scale,
        const float* __restrict__ cos_sum, const float* __restrict__ sin_sum,
        float* __restrict__ alpha,          // read for l_u, written for l_n
        float* __restrict__ mem_r, float* __restrict__ mem_i,
        const int* __restrict__ tokens, const float* __restrict__ emb,
        const float* __restrict__ collapse_w, const float* __restrict__ collapse_b,
        const float* __restrict__ w_trigger, const float* __restrict__ b_trigger,
        const float* __restrict__ trigger_bias,
        const float* __restrict__ sintab, const float* __restrict__ costab,
        float* __restrict__ xr_buf, float* __restrict__ xi_buf,
        float* __restrict__ xc,
        _Float16* __restrict__ Apack,
        int has_update, int l_u, int t_u, int write_apack,
        int do_evolve, int do_next, int l_n, int t_n) {
    __shared__ float cs_s[NOSC];
    __shared__ float ss_s[NOSC];
    __shared__ float xs[2 * DMODEL];
    __shared__ float red[DMODEL];

    int b = blockIdx.x;
    int d = threadIdx.x;

    float xr, xi;
    if (has_update) {
        cs_s[d]          = cos_sum[b * NOSC + d];
        cs_s[DMODEL + d] = cos_sum[b * NOSC + DMODEL + d];
        ss_s[d]          = sin_sum[b * NOSC + d];
        ss_s[DMODEL + d] = sin_sum[b * NOSC + DMODEL + d];
        __syncthreads();
        const float* orow = out_r + ((size_t)l_u * DMODEL + d) * NOSC;
        const float* irow = out_i + ((size_t)l_u * DMODEL + d) * NOSC;
        float ar = 0.f, ai = 0.f;
        #pragma unroll 8
        for (int n = 0; n < NOSC; ++n) { ar += cs_s[n] * orow[n]; ai += ss_s[n] * irow[n]; }
        float resr = ar / (1.f + expf(-ar));   // silu
        float resi = ai / (1.f + expf(-ai));
        float tphi = (float)t_u * PHI_F;
        xr = xr_buf[b * DMODEL + d];
        xi = xi_buf[b * DMODEL + d];
        float echor = 0.f, echoi = 0.f;
        #pragma unroll
        for (int h = 0; h < NHEAD; ++h) {
            float a = alpha[b * NHEAD + h];
            size_t midx = (((size_t)l_u * BSZ + b) * NHEAD + h) * DMODEL + d;
            float blr = a * xr + (1.f - a) * mem_r[midx];
            float bli = a * xi + (1.f - a) * mem_i[midx];
            int sbase = (l_u * NHEAD + h) * DMODEL + d;
            float wls = 1.f + fabsf(w_state[sbase]);
            float bs  = b_state[sbase];
            float s_sr, c_sr, s_si, c_si;
            lut_sincos(blr / wls + bs + tphi, sintab, costab, s_sr, c_sr);
            lut_sincos(bli / wls + bs + tphi, sintab, costab, s_si, c_si);
            float evr = c_sr * c_si - s_sr * s_si;
            float evi = c_sr * s_si + s_sr * c_si;
            mem_r[midx] = evr;
            mem_i[midx] = evi;
            echor += evr;
            echoi += evi;
        }
        float rs = res_scale[l_u], es = echo_scale[l_u];
        xr += rs * resr + es * echor;
        xi += rs * resi + es * echoi;
    } else {
        xr = xr_buf[b * DMODEL + d];
        xi = xi_buf[b * DMODEL + d];
    }

    if (write_apack) {
        size_t row = (size_t)b * TSEQ + t_u;   // logits output is (B, T, V)
        Apack[row * (2 * DMODEL) + d]          = (_Float16)xr;
        Apack[row * (2 * DMODEL) + DMODEL + d] = (_Float16)xi;
    }

    if (do_evolve) {   // state -> x for time t_n (state_r == xr)
        int tok   = tokens[b * TSEQ + t_n];
        float w   = emb[(size_t)tok * (2 * DMODEL) + d];
        float bb  = emb[(size_t)tok * (2 * DMODEL) + DMODEL + d];
        float wl  = 1.f + fabsf(w);
        float th  = xr / wl + bb + (float)t_n * PHI_F;
        float s, c;
        lut_sincos(th, sintab, costab, s, c);
        xr = c; xi = s;
    }

    xr_buf[b * DMODEL + d] = xr;
    xi_buf[b * DMODEL + d] = xi;

    if (do_next) {
        // ---- collapse: xc = concat(x_r,x_i) @ collapse_w[l_n].T + b ----
        xs[d]          = xr;
        xs[DMODEL + d] = xi;
        __syncthreads();
        const float* wrow = collapse_w + ((size_t)l_n * DMODEL + d) * (2 * DMODEL);
        float acc = collapse_b[l_n * DMODEL + d];
        #pragma unroll 8
        for (int j = 0; j < 2 * DMODEL; ++j) acc += xs[j] * wrow[j];
        xc[b * DMODEL + d] = acc;

        // ---- trigger -> alpha[b,h] for l_n, t_n ----
        float tphi_n = (float)t_n * PHI_F;
        #pragma unroll
        for (int h = 0; h < NHEAD; ++h) {
            int base = (l_n * NHEAD + h) * DMODEL + d;
            float wlt = 1.f + fabsf(w_trigger[base]);
            float bt  = b_trigger[base];
            float s_tr, c_tr, s_ti, c_ti;
            lut_sincos(xr / wlt + bt + tphi_n, sintab, costab, s_tr, c_tr);
            lut_sincos(xi / wlt + bt + tphi_n, sintab, costab, s_ti, c_ti);
            float trig_r = c_tr * c_ti - s_tr * s_ti;
            float trig_i = c_tr * s_ti + s_tr * c_ti;
            red[d] = trig_r * xr + trig_i * xi;
            __syncthreads();
            #pragma unroll
            for (int sft = 128; sft >= 1; sft >>= 1) {
                if (d < sft) red[d] += red[d + sft];
                __syncthreads();
            }
            if (d == 0) {
                float z = red[0] / 16.0f + trigger_bias[l_n * NHEAD + h]; // scale=sqrt(256)
                alpha[b * NHEAD + h] = 1.f / (1.f + expf(-z));
            }
            __syncthreads();
        }
    }
}

// ---------------------------------------------------------------------------
// Final logits GEMM with WMMA: D[m,n] = sum_k Apack[m,k] * Bpack[n,k]
// M = 2048 (b*T+t), N = 50257, K = 512 ([out_proj_r | out_proj_i] rows).
// Block tile 128x128, 8 waves, each wave 32(M) x 64(N) = 2x4 WMMA tiles.
// Double-buffered LDS staging; B read f32 from global (L2-resident),
// converted f16 on the fly; prefetch 2 chunks ahead on the B stream.
// ---------------------------------------------------------------------------
__global__ __launch_bounds__(256) void k_logits_wmma(
        const _Float16* __restrict__ Apack,
        const float*    __restrict__ out_proj_r,
        const float*    __restrict__ out_proj_i,
        float*          __restrict__ out) {
    __shared__ _Float16 A_s[2][128][32];
    __shared__ _Float16 B_s[2][128][32];

    int tid    = threadIdx.x;
    int blockN = blockIdx.x;
    int blockM = blockIdx.y;
    int lane   = tid & 31;
    int wave   = tid >> 5;
    int wm     = (wave & 3) * 32;   // wave M offset in block tile
    int wn     = (wave >> 2) * 64;  // wave N offset in block tile
    int lm     = lane & 15;
    int lh     = lane >> 4;

    v8f acc[2][4];
    #pragma unroll
    for (int mi = 0; mi < 2; ++mi)
        #pragma unroll
        for (int ni = 0; ni < 4; ++ni)
            #pragma unroll
            for (int j = 0; j < 8; ++j) acc[mi][ni][j] = 0.0f;

    int r = tid >> 1;   // staging row 0..127
    int p = tid & 1;    // staging half (16 elements each)
    int gn_stage = blockN * 128 + r;

    auto stage = [&](int kb, int buf) {
        // A: f16 128x32 chunk
        const uint4* srcA = (const uint4*)(Apack +
            ((size_t)(blockM * 128 + r)) * (2 * DMODEL) + kb + p * 16);
        uint4* dstA = (uint4*)&A_s[buf][r][p * 16];
        dstA[0] = srcA[0];
        dstA[1] = srcA[1];
        // B: f32 -> f16 128x32 chunk
        __align__(16) _Float16 tmp[16];
        if (gn_stage < VOCAB) {
            const float* bsrc = ((kb < DMODEL) ? out_proj_r : out_proj_i) +
                                (size_t)gn_stage * DMODEL + (kb & (DMODEL - 1)) + p * 16;
            #pragma unroll
            for (int j = 0; j < 16; ++j) tmp[j] = (_Float16)bsrc[j];
            if (kb + 64 < 2 * DMODEL) {   // prefetch 2 chunks ahead -> global_prefetch_b8
                const float* pre = ((kb + 64 < DMODEL) ? out_proj_r : out_proj_i) +
                                   (size_t)gn_stage * DMODEL + ((kb + 64) & (DMODEL - 1)) + p * 16;
                __builtin_prefetch((const void*)pre, 0, 1);
            }
        } else {
            #pragma unroll
            for (int j = 0; j < 16; ++j) tmp[j] = (_Float16)0.f;
        }
        uint4* dstB = (uint4*)&B_s[buf][r][p * 16];
        dstB[0] = ((const uint4*)tmp)[0];
        dstB[1] = ((const uint4*)tmp)[1];
    };

    stage(0, 0);
    __syncthreads();

    const int KITERS = (2 * DMODEL) / 32;   // 16
    for (int ki = 0; ki < KITERS; ++ki) {
        int cur = ki & 1;
        if (ki + 1 < KITERS) stage((ki + 1) * 32, cur ^ 1);

        // ---- fragments (contiguous k = 16*lh + s, same in A and B) ----
        v16h af[2];
        #pragma unroll
        for (int mi = 0; mi < 2; ++mi) {
            af[mi] = *(const v16h*)&A_s[cur][wm + mi * 16 + lm][lh * 16];
        }
        v16h bf[4];
        #pragma unroll
        for (int ni = 0; ni < 4; ++ni)
            bf[ni] = *(const v16h*)&B_s[cur][wn + ni * 16 + lm][lh * 16];

        #pragma unroll
        for (int mi = 0; mi < 2; ++mi)
            #pragma unroll
            for (int ni = 0; ni < 4; ++ni)
                acc[mi][ni] = __builtin_amdgcn_wmma_f32_16x16x32_f16(
                    false, af[mi], false, bf[ni], (short)0, acc[mi][ni], false, false);

        __syncthreads();
    }

    // ---- store D (lane = N, vgpr rr + 8*lh = M) ----
    #pragma unroll
    for (int mi = 0; mi < 2; ++mi) {
        #pragma unroll
        for (int ni = 0; ni < 4; ++ni) {
            int gn = blockN * 128 + wn + ni * 16 + lm;
            if (gn < VOCAB) {
                #pragma unroll
                for (int rr = 0; rr < 8; ++rr) {
                    int gm = blockM * 128 + wm + mi * 16 + 8 * lh + rr;
                    out[(size_t)gm * VOCAB + gn] = acc[mi][ni][rr];
                }
            }
        }
    }
}

// ---------------------------------------------------------------------------
extern "C" void kernel_launch(void* const* d_in, const int* in_sizes, int n_in,
                              void* d_out, int out_size, void* d_ws, size_t ws_size,
                              hipStream_t stream) {
    (void)in_sizes; (void)n_in; (void)out_size; (void)ws_size;

    const int*   tokens       = (const int*)  d_in[0];
    const float* emb          = (const float*)d_in[1];
    const float* collapse_w   = (const float*)d_in[2];
    const float* collapse_b   = (const float*)d_in[3];
    const float* Wm           = (const float*)d_in[4];
    const float* Bm           = (const float*)d_in[5];
    const float* attn_cos     = (const float*)d_in[6];
    const float* attn_sin     = (const float*)d_in[7];
    const float* out_r        = (const float*)d_in[8];
    const float* out_i        = (const float*)d_in[9];
    const float* w_trigger    = (const float*)d_in[10];
    const float* b_trigger    = (const float*)d_in[11];
    const float* w_state      = (const float*)d_in[12];
    const float* b_state      = (const float*)d_in[13];
    const float* trigger_bias = (const float*)d_in[14];
    const float* res_scale    = (const float*)d_in[15];
    const float* echo_scale   = (const float*)d_in[16];
    const float* out_proj_r   = (const float*)d_in[17];
    const float* out_proj_i   = (const float*)d_in[18];
    float* out = (float*)d_out;

    // ---- carve workspace (aligned to 256B) ----
    char*  wbase = (char*)d_ws;
    size_t off   = 0;
    auto carve = [&](size_t bytes) -> void* {
        size_t a = (off + 255) & ~(size_t)255;
        off = a + bytes;
        return (void*)(wbase + a);
    };
    float* sintab  = (float*)carve(LUTN * sizeof(float));
    float* costab  = (float*)carve(LUTN * sizeof(float));
    float* xr_buf  = (float*)carve(BSZ * DMODEL * sizeof(float));
    float* xi_buf  = (float*)carve(BSZ * DMODEL * sizeof(float));
    float* xc      = (float*)carve(BSZ * DMODEL * sizeof(float));
    float* cos_sum = (float*)carve(BSZ * NOSC * sizeof(float));
    float* sin_sum = (float*)carve(BSZ * NOSC * sizeof(float));
    float* alpha   = (float*)carve(BSZ * NHEAD * sizeof(float));
    float* mem_r   = (float*)carve((size_t)NLAYER * BSZ * NHEAD * DMODEL * sizeof(float));
    float* mem_i   = (float*)carve((size_t)NLAYER * BSZ * NHEAD * DMODEL * sizeof(float));
    _Float16* Apack = (_Float16*)carve((size_t)BSZ * TSEQ * (2 * DMODEL) * sizeof(_Float16));

    k_init<<<64, 256, 0, stream>>>(sintab, costab, xr_buf, xi_buf, mem_r, mem_i);

    #define LAUNCH_FUSED(hu, lu, tu, wa, ev, dn, ln, tn)                                \
        k_fused<<<BSZ, DMODEL, 0, stream>>>(out_r, out_i, w_state, b_state,             \
            res_scale, echo_scale, cos_sum, sin_sum, alpha, mem_r, mem_i,               \
            tokens, emb, collapse_w, collapse_b, w_trigger, b_trigger, trigger_bias,    \
            sintab, costab, xr_buf, xi_buf, xc, Apack, (hu), (lu), (tu), (wa), (ev),    \
            (dn), (ln), (tn))

    // prologue: evolve t=0 state, collapse+trigger for layer 0, t=0
    LAUNCH_FUSED(0, 0, 0, 0, 1, 1, 0, 0);

    for (int t = 0; t < TSEQ; ++t) {
        k_theta_attn<<<512, 256, 0, stream>>>(Wm, Bm, attn_cos, attn_sin,
                                              sintab, costab, xc, cos_sum, sin_sum, 0, t);
        // update layer 0 @ t, then collapse+trigger layer 1 @ t
        LAUNCH_FUSED(1, 0, t, 0, 0, 1, 1, t);
        k_theta_attn<<<512, 256, 0, stream>>>(Wm, Bm, attn_cos, attn_sin,
                                              sintab, costab, xc, cos_sum, sin_sum, 1, t);
        if (t + 1 < TSEQ) {
            // update layer 1 @ t (+Apack), evolve t+1, collapse+trigger layer 0 @ t+1
            LAUNCH_FUSED(1, 1, t, 1, 1, 1, 0, t + 1);
        } else {
            // epilogue: final update layer 1 @ T-1 (+Apack)
            LAUNCH_FUSED(1, 1, t, 1, 0, 0, 0, 0);
        }
    }
    #undef LAUNCH_FUSED

    dim3 grid((VOCAB + 127) / 128, (BSZ * TSEQ) / 128);
    k_logits_wmma<<<grid, 256, 0, stream>>>(Apack, out_proj_r, out_proj_i, out);
}
